// modnetSigmoidEncoder_75342316307140
// MI455X (gfx1250) — matrix operand order repaired and checked by reading, not compiled, not run on hardware
//
#include <hip/hip_runtime.h>
#include <hip/hip_bf16.h>
#include <math.h>

typedef _Float16 h8   __attribute__((ext_vector_type(8)));
typedef _Float16 v16h __attribute__((ext_vector_type(16)));
typedef float    v8f  __attribute__((ext_vector_type(8)));

#define B_    32
#define H_    40
#define W_    256
#define FREQ_ 40
#define C0_   32
#define H0_   38
#define W0_   254
#define C1_   64
#define H1_   36
#define W1_   252
#define SP1_  9072                 // H1_*W1_
#define CNN_  580608               // C1_*SP1_ = 1024*567
#define NT1_  567                  // SP1_/16
#define KCHUNK_ 1024               // divides CNN_ exactly -> no tail guards
#define NKB_  (CNN_ / KCHUNK_)     // 567
#define GST_  132                  // padded LDS row stride (16B-aligned rows: 528B)
#define PI_F  3.14159265358979323846f

#if defined(__has_builtin)
#if __has_builtin(__builtin_amdgcn_global_load_async_to_lds_b128) && \
    __has_builtin(__builtin_amdgcn_s_wait_asynccnt)
#define HAVE_ASYNC_LDS 1
#endif
#endif
#ifndef HAVE_ASYNC_LDS
#define HAVE_ASYNC_LDS 0
#endif

#if HAVE_ASYNC_LDS
typedef int v4i_ __attribute__((vector_size(16)));
typedef __attribute__((address_space(1))) v4i_* gptr4_;   // global int4*
typedef __attribute__((address_space(3))) v4i_* lptr4_;   // LDS int4* (32-bit offset)
__device__ __forceinline__ void async_copy16(const float* g, float* l) {
    // AS1 global pointer == flat VA; AS3 LDS pointer == low 32 bits of the
    // generic shared address.
    __builtin_amdgcn_global_load_async_to_lds_b128(
        (gptr4_)(__UINTPTR_TYPE__)g,
        (lptr4_)(unsigned int)(__UINTPTR_TYPE__)l,
        0, 0);
}
#endif

// ---------------------------------------------------------------- sin/cos tables
__global__ void k_tables(float* __restrict__ sinT, float* __restrict__ cosT) {
    int tid = blockIdx.x * blockDim.x + threadIdx.x;      // FREQ_*W_ = 10240
    if (tid >= FREQ_ * W_) return;
    int f = tid >> 8, w = tid & 255;
    float fs = 2.0f * (float)(f + 1);                     // (1/WIND)*linspace(1,40,40)
    float t  = 0.5f * (float)w / 255.0f;                  // linspace(0,0.5,256)
    float ph = 2.0f * PI_F * fs * t;
    sinT[tid] = sinf(ph);
    cosT[tid] = cosf(ph);
}

// ---------------------------------------------------------------- feats = 1x5 conv along W + bias
__global__ void k_feats(const float* __restrict__ in, const float* __restrict__ wf,
                        const float* __restrict__ bf, float* __restrict__ f0) {
    int tid = blockIdx.x * blockDim.x + threadIdx.x;      // B_*H_*W_ = 327680
    if (tid >= B_ * H_ * W_) return;
    int w  = tid & 255;
    int bh = tid >> 8;                                    // b*H_ + h
    float acc = bf[0];
#pragma unroll
    for (int k = 0; k < 5; k++) {
        int ww = w + k - 2;
        if (ww >= 0 && ww < W_) acc += wf[k] * in[(bh << 8) + ww];
    }
    f0[tid] = acc;
}

// ---------------------------------------------------------------- conv0: 1->32ch 3x3 VALID + relu, out f16 NHWC
__global__ void k_conv0(const float* __restrict__ in, const float* __restrict__ w0,
                        const float* __restrict__ b0, _Float16* __restrict__ x0h) {
    int tid = blockIdx.x * blockDim.x + threadIdx.x;      // B_*H0_*W0_ = 308864
    if (tid >= B_ * H0_ * W0_) return;
    int x  = tid % W0_;
    int t2 = tid / W0_;
    int y  = t2 % H0_;
    int b  = t2 / H0_;
    const float* ip = in + ((size_t)(b * H_ + y)) * W_ + x;
    float p[9];
#pragma unroll
    for (int dy = 0; dy < 3; dy++)
#pragma unroll
        for (int dx = 0; dx < 3; dx++) p[dy * 3 + dx] = ip[dy * W_ + dx];
    _Float16* op = x0h + (size_t)tid * C0_;               // [b][y][x][ci], ci contiguous
    for (int co = 0; co < C0_; co++) {
        float acc = b0[co];
        const float* wp = w0 + co * 9;
#pragma unroll
        for (int k = 0; k < 9; k++) acc = fmaf(wp[k], p[k], acc);
        acc = acc > 0.0f ? acc : 0.0f;
        op[co] = (_Float16)acc;
    }
}

// ---------------------------------------------------------------- repack w1 (OIHW f32) -> [co][tap][ci] f16
__global__ void k_w1pack(const float* __restrict__ w1, _Float16* __restrict__ w1h) {
    int tid = blockIdx.x * blockDim.x + threadIdx.x;      // C1_*9*C0_ = 18432
    if (tid >= C1_ * 9 * C0_) return;
    int ci  = tid & 31;
    int t2  = tid >> 5;
    int tap = t2 % 9;
    int co  = t2 / 9;
    w1h[tid] = (_Float16)w1[(co * C0_ + ci) * 9 + tap];
}

// ---------------------------------------------------------------- conv1 as implicit GEMM via WMMA f16->f32
// Per wave: one 16(co) x 16(spatial) output tile, K = 9 taps x 32 ci (one WMMA per tap).
__global__ void __launch_bounds__(256) k_conv1(const _Float16* __restrict__ x0h,
                                               const _Float16* __restrict__ w1h,
                                               const float* __restrict__ b1,
                                               float* __restrict__ x1) {
    int wave = threadIdx.x >> 5;
    int lane = threadIdx.x & 31;
    int tile = blockIdx.x * 8 + wave;                     // 0 .. 72575 (exact: 9072 blocks * 8)
    int nt = tile % NT1_;
    int t2 = tile / NT1_;
    int mt = t2 & 3;
    int b  = t2 >> 2;
    int lh = lane & 15;
    int hs = lane >> 4;                                   // half-wave select
    int co0 = mt * 16;
    int s   = nt * 16 + lh;                               // spatial column (always < 9072)
    int y = s / W1_, x = s % W1_;

    v8f c = {};
    const _Float16* bbase = x0h + (((size_t)(b * H0_ + y) * W0_ + x) * C0_ + hs * 8);
    const _Float16* abase = w1h + (size_t)(co0 + lh) * 9 * C0_ + hs * 8;

#pragma unroll
    for (int tap = 0; tap < 9; tap++) {
        int ky = tap / 3, kx = tap % 3;
        // A (16x32 f16): lane row M=co0+lh; K halves {hs*8..hs*8+7} U {16+hs*8..16+hs*8+7}
        const _Float16* ap = abase + tap * C0_;
        h8 a_lo = *(const h8*)(ap);
        h8 a_hi = *(const h8*)(ap + 16);
        v16h A = __builtin_shufflevector(a_lo, a_hi, 0, 1, 2, 3, 4, 5, 6, 7,
                                                     8, 9, 10, 11, 12, 13, 14, 15);
        // B (32x16 f16): lane column N=lh; K halves {hs*8..hs*8+7} U {16+hs*8..16+hs*8+7} (ci contiguous in NHWC x0)
        const _Float16* bp = bbase + (size_t)(ky * W0_ + kx) * C0_;
        h8 b_lo = *(const h8*)(bp);
        h8 b_hi = *(const h8*)(bp + 16);
        v16h Bm = __builtin_shufflevector(b_lo, b_hi, 0, 1, 2, 3, 4, 5, 6, 7,
                                                      8, 9, 10, 11, 12, 13, 14, 15);
        c = __builtin_amdgcn_wmma_f32_16x16x32_f16(false, A, false, Bm, (short)0, c, false, false);
    }

    // D (16x16 f32): VGPR r -> row co0 + hs*8 + r, col = lh; add bias + relu, store NCHW
    float* out = x1 + (size_t)b * CNN_ + s;
#pragma unroll
    for (int r = 0; r < 8; r++) {
        int co = co0 + hs * 8 + r;
        float v = c[r] + b1[co];
        out[(size_t)co * SP1_] = v > 0.0f ? v : 0.0f;
    }
}

// ---------------------------------------------------------------- gates GEMM: split-K, async-LDS double-buffered
__global__ void __launch_bounds__(256) k_gates_partial(const float* __restrict__ x1,
                                                       const float* __restrict__ wr,
                                                       float* __restrict__ partials) {
#if HAVE_ASYNC_LDS
    __shared__ __align__(16) float xs[2][32 * GST_];
    __shared__ __align__(16) float wsm[2][40 * GST_];
#else
    __shared__ __align__(16) float xs[1][32 * GST_];
    __shared__ __align__(16) float wsm[1][40 * GST_];
#endif
    int t  = threadIdx.x;
    int kb = blockIdx.x;
    int kbase0 = kb * KCHUNK_;
    float acc[5] = {0.f, 0.f, 0.f, 0.f, 0.f};
    int bidx[5], fidx[5];
#pragma unroll
    for (int j = 0; j < 5; j++) { int p = t + 256 * j; bidx[j] = p / 40; fidx[j] = p % 40; }

#if HAVE_ASYNC_LDS
    // stage one 128-wide K tile into LDS buffer `buf` entirely via async-to-LDS b128
    auto stage = [&](int buf, int kbase) {
#pragma unroll
        for (int i = 0; i < 4; i++) {                     // xs: 32*128 floats = 1024 b128 ops
            int idx = (t + 256 * i) * 4;
            int bb = idx >> 7, kk = idx & 127;
            async_copy16(x1 + (size_t)bb * CNN_ + kbase + kk, &xs[buf][bb * GST_ + kk]);
        }
#pragma unroll
        for (int i = 0; i < 5; i++) {                     // wsm: 40*128 floats = 1280 b128 ops
            int idx = (t + 256 * i) * 4;
            int ff = idx >> 7, kk = idx & 127;
            async_copy16(wr + (size_t)ff * CNN_ + kbase + kk, &wsm[buf][ff * GST_ + kk]);
        }
    };
    stage(0, kbase0);
    __builtin_amdgcn_s_wait_asynccnt(0);
    __syncthreads();
    for (int tl = 0; tl < KCHUNK_ / 128; tl++) {
        int buf = tl & 1;
        if (tl + 1 < KCHUNK_ / 128) stage(buf ^ 1, kbase0 + (tl + 1) * 128);
        const float* xp = xs[buf];
        const float* wp = wsm[buf];
        for (int kk = 0; kk < 128; kk++) {
#pragma unroll
            for (int j = 0; j < 5; j++)
                acc[j] = fmaf(xp[bidx[j] * GST_ + kk], wp[fidx[j] * GST_ + kk], acc[j]);
        }
        __builtin_amdgcn_s_wait_asynccnt(0);
        __syncthreads();
    }
#else
    for (int tl = 0; tl < KCHUNK_ / 128; tl++) {
        int kbase = kbase0 + tl * 128;
#pragma unroll
        for (int i = 0; i < 16; i++) {                    // stage x1[32][128]
            int idx = t + 256 * i;
            int bb = idx >> 7, kk = idx & 127;
            xs[0][bb * GST_ + kk] = x1[(size_t)bb * CNN_ + kbase + kk];
        }
#pragma unroll
        for (int i = 0; i < 20; i++) {                    // stage wr[40][128] (HBM stream)
            int idx = t + 256 * i;
            int ff = idx >> 7, kk = idx & 127;
            wsm[0][ff * GST_ + kk] = wr[(size_t)ff * CNN_ + kbase + kk];
            if (kbase + kk + 128 < CNN_)
                __builtin_prefetch(&wr[(size_t)ff * CNN_ + kbase + kk + 128], 0, 0);
        }
        __syncthreads();
        for (int kk = 0; kk < 128; kk++) {
#pragma unroll
            for (int j = 0; j < 5; j++)
                acc[j] = fmaf(xs[0][bidx[j] * GST_ + kk], wsm[0][fidx[j] * GST_ + kk], acc[j]);
        }
        __syncthreads();
    }
#endif
#pragma unroll
    for (int j = 0; j < 5; j++) partials[(size_t)kb * 1280 + t + 256 * j] = acc[j];
}

// ---------------------------------------------------------------- fixed-order reduce + bias + sigmoid
__global__ void k_gates_reduce(const float* __restrict__ partials, const float* __restrict__ br,
                               float* __restrict__ gates) {
    int tid = blockIdx.x * blockDim.x + threadIdx.x;      // 1280
    if (tid >= B_ * FREQ_) return;
    float s = 0.0f;
    for (int kb = 0; kb < NKB_; kb++) s += partials[(size_t)kb * 1280 + tid];
    int f = tid % FREQ_;
    s += br[f];
    gates[tid] = 1.0f / (1.0f + expf(-s));
}

// ---------------------------------------------------------------- weighted mean of gates*fs -> out[51200]
__global__ void k_wmean(const float* __restrict__ gates, float* __restrict__ out) {
    __shared__ float red[256];
    int t = threadIdx.x;
    float s = 0.0f;
    for (int i = t; i < B_ * FREQ_; i += 256) {
        int f = i % FREQ_;
        s += gates[i] * 2.0f * (float)(f + 1);
    }
    red[t] = s;
    __syncthreads();
    for (int o = 128; o > 0; o >>= 1) {
        if (t < o) red[t] += red[t + o];
        __syncthreads();
    }
    if (t == 0) out[B_ * FREQ_ * H_] = red[0] / (float)(B_ * FREQ_);
}

// ---------------------------------------------------------------- Fourier mags * gates -> mods
__global__ void k_mods(const float* __restrict__ f0, const float* __restrict__ sinT,
                       const float* __restrict__ cosT, const float* __restrict__ gates,
                       float* __restrict__ out) {
    int tid = blockIdx.x * blockDim.x + threadIdx.x;      // B_*FREQ_*H_ = 51200
    if (tid >= B_ * FREQ_ * H_) return;
    int h  = tid % H_;
    int t2 = tid / H_;
    int f  = t2 % FREQ_;
    int b  = t2 / FREQ_;
    const float* fr = f0 + (size_t)(b * H_ + h) * W_;
    const float* sp = sinT + f * W_;
    const float* cp = cosT + f * W_;
    float s = 0.0f, c = 0.0f;
    for (int w = 0; w < W_; w++) {
        float v = fr[w];
        s = fmaf(sp[w], v, s);
        c = fmaf(cp[w], v, c);
    }
    float mag = sqrtf(s * s + c * c) * (1.0f / (float)W_);
    out[tid] = mag * gates[b * FREQ_ + f];                // layout b*1600 + f*40 + h
}

// ---------------------------------------------------------------- launch
extern "C" void kernel_launch(void* const* d_in, const int* in_sizes, int n_in,
                              void* d_out, int out_size, void* d_ws, size_t ws_size,
                              hipStream_t stream) {
    (void)in_sizes; (void)n_in; (void)out_size; (void)ws_size;
    const float* in = (const float*)d_in[0];
    const float* w0 = (const float*)d_in[1];
    const float* b0 = (const float*)d_in[2];
    const float* w1 = (const float*)d_in[3];
    const float* b1 = (const float*)d_in[4];
    const float* wf = (const float*)d_in[5];
    const float* bf = (const float*)d_in[6];
    const float* wr = (const float*)d_in[7];
    const float* br = (const float*)d_in[8];
    float* out = (float*)d_out;

    char* ws = (char*)d_ws;
    size_t off = 0;
    auto alloc = [&](size_t bytes) -> char* {
        char* p = ws + off;
        off += (bytes + 255) & ~(size_t)255;
        return p;
    };
    float*    f0       = (float*)alloc(sizeof(float) * B_ * H_ * W_);
    float*    sinT     = (float*)alloc(sizeof(float) * FREQ_ * W_);
    float*    cosT     = (float*)alloc(sizeof(float) * FREQ_ * W_);
    _Float16* w1h      = (_Float16*)alloc(sizeof(_Float16) * C1_ * 9 * C0_);
    _Float16* x0h      = (_Float16*)alloc(sizeof(_Float16) * (size_t)B_ * H0_ * W0_ * C0_);
    float*    x1       = (float*)alloc(sizeof(float) * (size_t)B_ * CNN_);
    float*    partials = (float*)alloc(sizeof(float) * (size_t)NKB_ * 1280);
    float*    gates    = (float*)alloc(sizeof(float) * B_ * FREQ_);

    k_tables<<<(FREQ_ * W_ + 255) / 256, 256, 0, stream>>>(sinT, cosT);
    k_feats<<<(B_ * H_ * W_ + 255) / 256, 256, 0, stream>>>(in, wf, bf, f0);
    k_conv0<<<(B_ * H0_ * W0_ + 255) / 256, 256, 0, stream>>>(in, w0, b0, x0h);
    k_w1pack<<<(C1_ * 9 * C0_ + 255) / 256, 256, 0, stream>>>(w1, w1h);
    k_conv1<<<SP1_, 256, 0, stream>>>(x0h, w1h, b1, x1);           // 9072 blocks * 8 wave-tiles
    k_gates_partial<<<NKB_, 256, 0, stream>>>(x1, wr, partials);
    k_gates_reduce<<<5, 256, 0, stream>>>(partials, br, gates);
    k_wmean<<<1, 256, 0, stream>>>(gates, out);
    k_mods<<<(B_ * FREQ_ * H_ + 255) / 256, 256, 0, stream>>>(f0, sinT, cosT, gates, out);
}
